// SlotNamePredictor_19670950216374
// MI455X (gfx1250) — matrix operand order repaired and checked by reading, not compiled, not run on hardware
//
#include <hip/hip_runtime.h>
#include <hip/hip_bf16.h>
#include <stdint.h>

// ---------------------------------------------------------------------------
// SlotNamePredictor: span-gather + bi-LSTM + masked pooling + slot logits
// CDNA5 (gfx1250): bf16 WMMA 16x16x32 for the input projection GEMM
// (double-buffered LDS B tiles via GLOBAL_LOAD_ASYNC_TO_LDS_B128, software-
// pipelined A fragments) and the per-step recurrent GEMM.
// ---------------------------------------------------------------------------

#define B_    64
#define S_    1024
#define D_    800      // input dim, = 25 * 32 (exact K-steps)
#define MS_   16
#define T_    32
#define H_    200
#define H4_   800      // 4*H, = 50 * 16 N-tiles
#define NSEQ_ 1024     // B*MS
#define NS_   12
#define HP_   224      // H padded to 7*32 for recurrent K

typedef __attribute__((ext_vector_type(16))) __bf16 v16bf;
typedef __attribute__((ext_vector_type(2)))  __bf16 v2bf;
typedef __attribute__((ext_vector_type(8)))  float  v8f;

__device__ __forceinline__ v16bf ldv16(const __bf16* p) {
  return *reinterpret_cast<const v16bf*>(p);
}

__device__ __forceinline__ v8f wmma_bf16(v16bf a, v16bf b, v8f c) {
  // D = A(16x32) * B(32x16) + C(16x16 f32)
  return __builtin_amdgcn_wmma_f32_16x16x32_bf16(
      /*neg_a=*/false, a, /*neg_b=*/false, b,
      /*c_mod=*/(short)0, c, /*reuse_a=*/false, /*reuse_b=*/false);
}

__device__ __forceinline__ float sigf(float x) {
  return 1.0f / (1.0f + __expf(-x));
}

// Async global->LDS copy of 16 bytes (VGLOBAL GLOBAL_LOAD_ASYNC_TO_LDS_B128).
// lds generic pointer: low 32 bits are the LDS byte offset (aperture rule).
__device__ __forceinline__ void async_g2l_b128(const void* gsrc, void* ldst) {
  uint32_t l = (uint32_t)(uintptr_t)ldst;
  uint64_t g = (uint64_t)(uintptr_t)gsrc;
  asm volatile("global_load_async_to_lds_b128 %0, %1, off"
               :: "v"(l), "v"(g) : "memory");
}

__device__ __forceinline__ void wait_asynccnt0() {
  asm volatile("s_wait_asynccnt 0x0" ::: "memory");
}

// ---------------------------------------------------------------------------
// 1) Gather span tokens -> Xbf[NSEQ][T][D] (bf16)
// ---------------------------------------------------------------------------
__global__ __launch_bounds__(128)
void snp_gather(const float* __restrict__ hid,
                const int* __restrict__ starts,
                void* __restrict__ xbf_) {
  __bf16* xbf = reinterpret_cast<__bf16*>(xbf_);
  int row = blockIdx.x;              // 0 .. NSEQ_*T_-1
  int seq = row >> 5;                // T_ == 32
  int t   = row & 31;
  int b   = seq >> 4;                // MS_ == 16
  int srow = starts[seq] + t;
  srow = srow < 0 ? 0 : (srow > S_ - 1 ? S_ - 1 : srow);
  const float* src = hid + ((size_t)b * S_ + srow) * D_;
  __bf16* dst = xbf + (size_t)row * D_;
  for (int p = threadIdx.x; p < D_ / 2; p += blockDim.x) {
    float2 v = reinterpret_cast<const float2*>(src)[p];
    v2bf o;
    o.x = (__bf16)v.x;
    o.y = (__bf16)v.y;
    reinterpret_cast<v2bf*>(dst)[p] = o;
  }
}

// ---------------------------------------------------------------------------
// 2) Weight conversion (per direction):
//    Wb[k][n]   = bf16(Wih[n][k])            K-major for B fragments
//    Whhb[k][n] = bf16(k<H ? Whh[n][k] : 0)  padded to HP_ rows
//    bias[n]    = bih[n] + bhh[n]
// ---------------------------------------------------------------------------
__global__ __launch_bounds__(256)
void snp_prep(const float* __restrict__ Wih, const float* __restrict__ Whh,
              const float* __restrict__ bih, const float* __restrict__ bhh,
              void* __restrict__ Wb_, void* __restrict__ Whhb_,
              float* __restrict__ bias) {
  __bf16* Wb   = reinterpret_cast<__bf16*>(Wb_);
  __bf16* Whhb = reinterpret_cast<__bf16*>(Whhb_);
  int e = blockIdx.x * blockDim.x + threadIdx.x;
  if (e < D_ * H4_) {
    int k = e / H4_, n = e - k * H4_;
    Wb[e] = (__bf16)Wih[n * D_ + k];
  }
  if (e < HP_ * H4_) {
    int k = e / H4_, n = e - k * H4_;
    Whhb[e] = (__bf16)((k < H_) ? Whh[n * H_ + k] : 0.0f);
  }
  if (e < H4_) bias[e] = bih[e] + bhh[e];
}

// ---------------------------------------------------------------------------
// 3) Input projection GEMM: Z[32768][800] = Xbf @ Wb + bias
//    WG tile 128x80: 8 waves x (16 M x 80 N), K loop 25 x 32.
//    Double-buffered B tile filled by async global->LDS DMA; A fragment
//    software-pipelined one iteration ahead; one wait+barrier per iter.
// ---------------------------------------------------------------------------
__global__ __launch_bounds__(256)
void snp_gemm_in(const void* __restrict__ X_, const void* __restrict__ Wb_,
                 const float* __restrict__ bias, float* __restrict__ Z) {
  const __bf16* X  = reinterpret_cast<const __bf16*>(X_);
  const __bf16* Wb = reinterpret_cast<const __bf16*>(Wb_);
  __shared__ __align__(32) __bf16 sB[2][32 * 80];

  int tid  = threadIdx.x;
  int wave = tid >> 5;
  int lane = tid & 31;
  int m0 = blockIdx.x * 128 + wave * 16;   // wave's 16 M rows
  int n0 = blockIdx.y * 80;                // WG's 80 N cols

  v8f acc[5] = {};

  // A fragment address: lane&15 -> M row, lane&16 -> K half (contiguous k = 16h + s, same order as B)
  int arow = m0 + (lane & 15);
  int koff = lane & 16;
  const __bf16* Aptr = X + (size_t)arow * D_ + koff;

  // staging geometry: 32 rows x 10 x 16B chunks = 320 async b128 copies
  int e0 = tid;               // chunk 0..255
  int r0 = e0 / 10, c0 = e0 - r0 * 10;
  int e1 = tid + 256;         // chunk 256..319 (only tid < 64)
  int r1 = e1 / 10, c1 = e1 - r1 * 10;

  // prologue: stage kt = 0 into buffer 0, and pipeline the first A fragment
  {
    const __bf16* Bsrc = Wb + n0;
    async_g2l_b128(Bsrc + (size_t)r0 * H4_ + c0 * 8, &sB[0][r0 * 80 + c0 * 8]);
    if (e1 < 320)
      async_g2l_b128(Bsrc + (size_t)r1 * H4_ + c1 * 8, &sB[0][r1 * 80 + c1 * 8]);
  }
  v16bf a_cur = ldv16(Aptr);
  wait_asynccnt0();
  __syncthreads();

  for (int kt = 0; kt < 25; ++kt) {
    int cur = kt & 1;
    // fire-and-forget stage of next B tile; prefetch A two tiles ahead;
    // software-pipeline next A fragment into registers.
    v16bf a_next;
    if (kt < 24) {
      const __bf16* Bsrc = Wb + (size_t)((kt + 1) * 32) * H4_ + n0;
      __bf16* dstb = sB[cur ^ 1];
      async_g2l_b128(Bsrc + (size_t)r0 * H4_ + c0 * 8, dstb + r0 * 80 + c0 * 8);
      if (e1 < 320)
        async_g2l_b128(Bsrc + (size_t)r1 * H4_ + c1 * 8, dstb + r1 * 80 + c1 * 8);
      a_next = ldv16(Aptr + (kt + 1) * 32);
      if (kt < 23) __builtin_prefetch(Aptr + (kt + 2) * 32, 0, 3);
    }

#pragma unroll
    for (int nt = 0; nt < 5; ++nt) {
      // B fragment: lane&15 -> N col, lane&16 -> K half, 16 contiguous K
      v16bf bfrag;
#pragma unroll
      for (int s = 0; s < 16; ++s)
        bfrag[s] = sB[cur][(koff + s) * 80 + nt * 16 + (lane & 15)];
      acc[nt] = wmma_bf16(a_cur, bfrag, acc[nt]);
    }
    wait_asynccnt0();
    __syncthreads();
    if (kt < 24) a_cur = a_next;
  }

  // epilogue: C layout -> lane&15 = N col, vgpr r = M row (+8 for lanes>=16)
  int m8   = (lane >> 4) * 8;
  int ncol = lane & 15;
#pragma unroll
  for (int nt = 0; nt < 5; ++nt) {
    float bv = bias[n0 + nt * 16 + ncol];
    float* Cp = Z + (size_t)(m0 + m8) * H4_ + n0 + nt * 16 + ncol;
#pragma unroll
    for (int r = 0; r < 8; ++r)
      Cp[(size_t)r * H4_] = acc[nt][r] + bv;
  }
}

// ---------------------------------------------------------------------------
// 4) Recurrent scan, one direction. 64 WGs x 16 sequences, 8 waves.
//    Per step: zbuf[16][800] = h_bf[16][224] @ Whhb[224][800] via WMMA
//    (A fragments hoisted across N-tiles), then masked LSTM cell update.
// ---------------------------------------------------------------------------
__global__ __launch_bounds__(256)
void snp_recur(const void* __restrict__ Whhb_, const float* __restrict__ Z,
               const int* __restrict__ lens, float* __restrict__ feats,
               int dir) {
  const __bf16* Whhb = reinterpret_cast<const __bf16*>(Whhb_);
  __shared__ __align__(32) __bf16 s_h[16 * HP_];  //  7,168 B  h (bf16, padded K)
  __shared__ float s_z[16 * H4_];                 // 51,200 B  h@Whh^T preacts
  __shared__ float s_c[16 * H_];                  // 12,800 B  cell state
  __shared__ float s_f[16 * H_];                  // 12,800 B  masked h sum

  int tid  = threadIdx.x;
  int wave = tid >> 5;
  int lane = tid & 31;
  int seqbase = blockIdx.x * 16;

  for (int e = tid; e < 16 * HP_; e += 256) s_h[e] = (__bf16)0.0f;
  for (int e = tid; e < 16 * H_; e += 256) { s_c[e] = 0.0f; s_f[e] = 0.0f; }
  __syncthreads();

  int arow = lane & 15;
  int koff = lane & 16;
  int m8   = (lane >> 4) * 8;
  int ncol = lane & 15;

  for (int step = 0; step < T_; ++step) {
    // --- matrix phase: load the wave's 7 A fragments once, sweep N-tiles ---
    v16bf afrag[7];
#pragma unroll
    for (int kt = 0; kt < 7; ++kt)
      afrag[kt] = ldv16(&s_h[arow * HP_ + kt * 32 + koff]);

    for (int nt = wave; nt < 50; nt += 8) {
      v8f acc = {};
#pragma unroll
      for (int kt = 0; kt < 7; ++kt) {
        v16bf b;
#pragma unroll
        for (int s = 0; s < 16; ++s)
          b[s] = Whhb[(size_t)(kt * 32 + koff + s) * H4_ + nt * 16 + ncol];
        acc = wmma_bf16(afrag[kt], b, acc);
      }
#pragma unroll
      for (int r = 0; r < 8; ++r)
        s_z[(m8 + r) * H4_ + nt * 16 + ncol] = acc[r];
    }
    __syncthreads();

    // --- elementwise LSTM cell: 16 seq x 200 units over 256 threads ---
    int tmem = dir ? (T_ - 1 - step) : step;
    for (int e = tid; e < 16 * H_; e += 256) {
      int s = e / H_, u = e - s * H_;
      const float* zrow = Z + ((size_t)(seqbase + s) * T_ + tmem) * H4_;
      float zi = s_z[s * H4_ + u]            + zrow[u];
      float zf = s_z[s * H4_ + H_ + u]       + zrow[H_ + u];
      float zg = s_z[s * H4_ + 2 * H_ + u]   + zrow[2 * H_ + u];
      float zo = s_z[s * H4_ + 3 * H_ + u]   + zrow[3 * H_ + u];
      float ig = sigf(zi), fg = sigf(zf), og = sigf(zo);
      float gg = tanhf(zg);
      float c_old = s_c[e];
      float c_new = fg * c_old + ig * gg;
      float h_new = og * tanhf(c_new);
      bool  mt = tmem < lens[seqbase + s];
      float h = mt ? h_new : (float)s_h[s * HP_ + u];
      if (mt) { s_c[e] = c_new; s_f[e] += h_new; }
      s_h[s * HP_ + u] = (__bf16)h;
    }
    __syncthreads();
  }

  for (int e = tid; e < 16 * H_; e += 256) {
    int s = e / H_, u = e - s * H_;
    feats[(size_t)(seqbase + s) * (2 * H_) + dir * H_ + u] = s_f[e];
  }
}

// ---------------------------------------------------------------------------
// 5) logits[b][m][n] = slot_mask[b][m] * sum_h feats[bm][h] * slot_embs[h][n]
// ---------------------------------------------------------------------------
__global__ __launch_bounds__(32)
void snp_logits(const float* __restrict__ feats, const float* __restrict__ emb,
                const float* __restrict__ smask, float* __restrict__ out) {
  int seq = blockIdx.x;     // 0..NSEQ_-1
  int n = threadIdx.x;
  if (n >= NS_) return;
  const float* f = feats + (size_t)seq * (2 * H_);
  float acc = 0.0f;
  for (int h = 0; h < 2 * H_; ++h) acc += f[h] * emb[h * NS_ + n];
  out[(size_t)seq * NS_ + n] = acc * smask[seq];
}

// ---------------------------------------------------------------------------
extern "C" void kernel_launch(void* const* d_in, const int* in_sizes, int n_in,
                              void* d_out, int out_size, void* d_ws, size_t ws_size,
                              hipStream_t stream) {
  const float* hid    = (const float*)d_in[0];
  const float* Wih[2] = {(const float*)d_in[1], (const float*)d_in[5]};
  const float* Whh[2] = {(const float*)d_in[2], (const float*)d_in[6]};
  const float* bih[2] = {(const float*)d_in[3], (const float*)d_in[7]};
  const float* bhh[2] = {(const float*)d_in[4], (const float*)d_in[8]};
  const float* emb    = (const float*)d_in[9];
  const int*   starts = (const int*)d_in[10];
  const int*   lens   = (const int*)d_in[11];
  const float* smask  = (const float*)d_in[12];
  float* out = (float*)d_out;

  char* w = (char*)d_ws;
  size_t off = 0;
  auto alignup = [](size_t x) { return (x + 511) & ~(size_t)511; };
  void*  Xbf  = w + off; off = alignup(off + (size_t)NSEQ_ * T_ * D_ * 2);
  void*  Wb   = w + off; off = alignup(off + (size_t)D_ * H4_ * 2);
  void*  Whhb = w + off; off = alignup(off + (size_t)HP_ * H4_ * 2);
  float* bias = (float*)(w + off); off = alignup(off + (size_t)H4_ * 4);
  float* Z    = (float*)(w + off); off = alignup(off + (size_t)NSEQ_ * T_ * H4_ * 4);
  float* feats= (float*)(w + off); off = alignup(off + (size_t)NSEQ_ * 2 * H_ * 4);
  (void)ws_size; (void)in_sizes; (void)n_in; (void)out_size;

  // Gather spans once (input projections share X across directions).
  snp_gather<<<NSEQ_ * T_, 128, 0, stream>>>(hid, starts, Xbf);

  for (int dir = 0; dir < 2; ++dir) {
    snp_prep<<<(D_ * H4_ + 255) / 256, 256, 0, stream>>>(
        Wih[dir], Whh[dir], bih[dir], bhh[dir], Wb, Whhb, bias);
    snp_gemm_in<<<dim3((NSEQ_ * T_) / 128, H4_ / 80), 256, 0, stream>>>(
        Xbf, Wb, bias, Z);
    snp_recur<<<NSEQ_ / 16, 256, 0, stream>>>(Whhb, Z, lens, feats, dir);
  }

  snp_logits<<<NSEQ_, 32, 0, stream>>>(feats, emb, smask, out);
}
